// Block_35347580846334
// MI455X (gfx1250) — matrix operand order repaired and checked by reading, not compiled, not run on hardware
//
#include <hip/hip_runtime.h>
#include <hip/hip_bf16.h>

typedef __attribute__((ext_vector_type(16))) _Float16 v16h;
typedef __attribute__((ext_vector_type(8)))  float    v8f;
typedef __attribute__((ext_vector_type(4)))  unsigned int v4u;
typedef __attribute__((ext_vector_type(4)))  unsigned int u32x4;
typedef __attribute__((ext_vector_type(8)))  int      i32x8;
typedef __attribute__((ext_vector_type(4)))  int      i32x4;

union Frag16 { v16h h; v4u q[2]; };

__device__ __forceinline__ v8f wmma_f16(v16h a, v16h b, v8f c) {
  // 8 args: (neg_a, A, neg_b, B, c_mod, C, reuse_a, reuse_b)
  return __builtin_amdgcn_wmma_f32_16x16x32_f16(false, a, false, b, (short)0, c,
                                                false, false);
}

// TDM: issue a 2D tile load (rows x cols of f16, row stride in elements)
// from global memory into LDS. D# packing per CDNA5 ISA
// (group0 = control/lds/global addr, group1 = dims/tile/stride, 2D so
// groups 2 and 3 are zero). data_size code 1 = 2 bytes.
// This toolchain's builtin is the 6-arg form:
//   (u32x4 g0, i32x8 g1, i32x4 g2, i32x4 g3, i32x8 extra, i32 cpol)
__device__ __forceinline__ void tdm_load_2d_f16(unsigned lds_addr,
                                                unsigned long long gaddr,
                                                int tensor_d0, int tensor_d1,
                                                int tile_d0, int tile_d1,
                                                long long stride0_elems) {
  u32x4 g0;
  g0.x = 1u;                                   // count=1, user mode
  g0.y = lds_addr;                             // LDS byte offset
  g0.z = (unsigned)(gaddr & 0xffffffffu);      // global_addr[31:0]
  g0.w = (unsigned)((gaddr >> 32) & 0x01ffffffu) | 0x80000000u;  // type=2

  const unsigned d0 = (unsigned)tensor_d0, d1 = (unsigned)tensor_d1;
  const unsigned long long s0 = (unsigned long long)stride0_elems;
  i32x8 g1;
  g1[0] = (int)(1u << 16);                     // wg_mask=0, data_size=1 (2B)
  g1[1] = (int)((d0 & 0xffffu) << 16);         // tensor_dim0[15:0]
  g1[2] = (int)((d0 >> 16) | ((d1 & 0xffffu) << 16));
  g1[3] = (int)((d1 >> 16) | ((unsigned)tile_d0 << 16));  // tile_dim0
  g1[4] = (int)((unsigned)tile_d1 & 0xffffu);  // tile_dim1 (tile_dim2=0)
  g1[5] = (int)(unsigned)(s0 & 0xffffffffull); // tensor_dim0_stride[31:0]
  g1[6] = (int)(unsigned)((s0 >> 32) & 0xffffull);
  g1[7] = 0;
  const i32x4 zz4 = {0, 0, 0, 0};
  const i32x8 zz8 = {0, 0, 0, 0, 0, 0, 0, 0};
  __builtin_amdgcn_tensor_load_to_lds(g0, g1, zz4, zz4, zz8, 0);
}

// ---------------------------------------------------------------------------
// f32 -> f16 conversion (weights)
// ---------------------------------------------------------------------------
__global__ __launch_bounds__(256) void cvt_kernel(const float* __restrict__ in,
                                                  _Float16* __restrict__ out,
                                                  int n) {
  int i = blockIdx.x * 256 + threadIdx.x;
  if (i < n) out[i] = (_Float16)in[i];
}

// ---------------------------------------------------------------------------
// LayerNorm: one block per row of 768, writes f16
// ---------------------------------------------------------------------------
__global__ __launch_bounds__(256) void ln_kernel(const float* __restrict__ x,
                                                 const float* __restrict__ g,
                                                 const float* __restrict__ bta,
                                                 _Float16* __restrict__ out) {
  __shared__ float red[256];
  const int row = blockIdx.x, tid = threadIdx.x;
  const float* xr = x + (size_t)row * 768;
  float v0 = xr[tid], v1 = xr[tid + 256], v2 = xr[tid + 512];
  red[tid] = v0 + v1 + v2;
  __syncthreads();
  for (int o = 128; o > 0; o >>= 1) {
    if (tid < o) red[tid] += red[tid + o];
    __syncthreads();
  }
  float mu = red[0] * (1.0f / 768.0f);
  __syncthreads();
  float d0 = v0 - mu, d1 = v1 - mu, d2 = v2 - mu;
  red[tid] = d0 * d0 + d1 * d1 + d2 * d2;
  __syncthreads();
  for (int o = 128; o > 0; o >>= 1) {
    if (tid < o) red[tid] += red[tid + o];
    __syncthreads();
  }
  float rstd = rsqrtf(red[0] * (1.0f / 768.0f) + 1e-5f);
  _Float16* orow = out + (size_t)row * 768;
  orow[tid]       = (_Float16)(d0 * rstd * g[tid]       + bta[tid]);
  orow[tid + 256] = (_Float16)(d1 * rstd * g[tid + 256] + bta[tid + 256]);
  orow[tid + 512] = (_Float16)(d2 * rstd * g[tid + 512] + bta[tid + 512]);
}

// ---------------------------------------------------------------------------
// WMMA GEMM: C[M,N] = A[M,K](f16,row) @ B[K,N](f16,row) + bias, epilogues:
//   EPI 0: f16 out
//   EPI 1: f16 out, per-head transposed V layout [B*H, 64, 1024]
//   EPI 2: exact-GELU, f16 out
//   EPI 3: + residual, f32 out
// A tile is staged via the Tensor Data Mover (async, TENSORcnt); B tile is
// staged transposed (n-major) through VGPRs for contiguous B-fragment loads.
// ---------------------------------------------------------------------------
template <int EPI>
__global__ __launch_bounds__(256) void gemm_kernel(
    const _Float16* __restrict__ A, const _Float16* __restrict__ B,
    const float* __restrict__ bias, const float* __restrict__ res,
    _Float16* __restrict__ out16, float* __restrict__ out32,
    int M, int N, int K) {
  __shared__ __align__(16) _Float16 As[128][32];  // k-contiguous (TDM dest)
  __shared__ __align__(16) _Float16 Bs[128][32];  // n-major (B^T), k-contiguous

  const int tid = threadIdx.x;
  const int lane = tid & 31, wave = tid >> 5;
  const int wm = wave >> 1, wn = wave & 1;
  const int m0 = blockIdx.y * 128, n0 = blockIdx.x * 128;
  const int lh = lane & 15;                 // fragment row/col within 16
  const int kg = (lane >> 4) * 8;           // A-frag K group base
  const int kb = (lane >> 4) * 16;          // unused; B frags use kg

  const unsigned as_lds = (unsigned)(uintptr_t)&As[0][0];  // LDS byte offset
  const unsigned long long a_base =
      (unsigned long long)(uintptr_t)A + (unsigned long long)m0 * K * 2;

  v8f acc[2][4];
#pragma unroll
  for (int i = 0; i < 2; ++i)
#pragma unroll
    for (int j = 0; j < 4; ++j) acc[i][j] = (v8f){};

  for (int k0 = 0; k0 < K; k0 += 32) {
    // ---- A tile: 128x32 f16, row stride K -> LDS via TDM (wave 0 only;
    //      TDM ignores EXEC, one issue per workgroup) ----
    if (wave == 0) {
      tdm_load_2d_f16(as_lds, a_base + (unsigned long long)k0 * 2,
                      K, M, 32, 128, (long long)K);
    }
    // ---- B tile: 32x128 -> transpose into Bs[n][k] ----
#pragma unroll
    for (int s = 0; s < 2; ++s) {
      const int seg = tid * 2 + s;  // 0..511
      const int kr = seg >> 4, np = (seg & 15) * 8;
      union { v4u u; _Float16 e[8]; } t;
      t.u = *(const v4u*)(B + (size_t)(k0 + kr) * N + n0 + np);
#pragma unroll
      for (int i = 0; i < 8; ++i) Bs[np + i][kr] = t.e[i];
    }
    if (k0 + 32 < K) {  // prefetch next B K-rows (global_prefetch_b8)
      __builtin_prefetch(B + (size_t)(k0 + 32 + (tid & 31)) * N + n0, 0, 0);
    }
    if (wave == 0) __builtin_amdgcn_s_wait_tensorcnt(0);
    __syncthreads();

    // ---- fragments + 8 WMMAs per wave ----
    Frag16 af[2], bf[4];
#pragma unroll
    for (int i = 0; i < 2; ++i) {
      const int row = wm * 32 + i * 16 + lh;
      af[i].q[0] = *(const v4u*)(&As[row][kg]);
      af[i].q[1] = *(const v4u*)(&As[row][kg + 16]);
    }
#pragma unroll
    for (int j = 0; j < 4; ++j) {
      const int col = wn * 64 + j * 16 + lh;
      bf[j].q[0] = *(const v4u*)(&Bs[col][kg]);
      bf[j].q[1] = *(const v4u*)(&Bs[col][kg + 16]);
    }
#pragma unroll
    for (int i = 0; i < 2; ++i)
#pragma unroll
      for (int j = 0; j < 4; ++j)
        acc[i][j] = wmma_f16(af[i].h, bf[j].h, acc[i][j]);
    __syncthreads();
  }

  // ---- epilogue ----
#pragma unroll
  for (int i = 0; i < 2; ++i)
#pragma unroll
    for (int j = 0; j < 4; ++j) {
      const int colg = n0 + wn * 64 + j * 16 + lh;
      const float bv = bias[colg];
#pragma unroll
      for (int r = 0; r < 8; ++r) {
        const int rowg = m0 + wm * 32 + i * 16 + r + (lane >> 4) * 8;
        float v = acc[i][j][r] + bv;
        if (EPI == 0) {
          out16[(size_t)rowg * N + colg] = (_Float16)v;
        } else if (EPI == 1) {
          const int bidx = rowg >> 10, ntok = rowg & 1023;
          const int hh = colg >> 6, dd = colg & 63;
          out16[(((size_t)(bidx * 12 + hh)) * 64 + dd) * 1024 + ntok] =
              (_Float16)v;
        } else if (EPI == 2) {
          v = 0.5f * v * (1.0f + erff(v * 0.70710678118654752f));
          out16[(size_t)rowg * N + colg] = (_Float16)v;
        } else {
          out32[(size_t)rowg * N + colg] = v + res[(size_t)rowg * N + colg];
        }
      }
    }
}

// ---------------------------------------------------------------------------
// Flash attention: grid = B*H*(1024/128) blocks, 8 waves, 16 q-rows per wave.
// q,k: [16384,768] f16 (head slice h*64). vt: [B*H,64,1024] f16 (pre-transposed).
// ---------------------------------------------------------------------------
__global__ __launch_bounds__(256) void attn_kernel(
    const _Float16* __restrict__ q, const _Float16* __restrict__ k,
    const _Float16* __restrict__ vt, _Float16* __restrict__ ctx) {
  __shared__ __align__(16) _Float16 pbuf[8][16][32];  // per-wave P tile
  const int tid = threadIdx.x, lane = tid & 31, wave = tid >> 5;
  const int qt = blockIdx.x & 7, bh = blockIdx.x >> 3;
  const int b = bh / 12, h = bh % 12;
  const int lh = lane & 15, kg = (lane >> 4) * 8, kb = (lane >> 4) * 16;
  const int q0 = qt * 128 + wave * 16;

  const _Float16* qp = q + (size_t)b * 1024 * 768 + h * 64;
  const _Float16* kp = k + (size_t)b * 1024 * 768 + h * 64;
  const _Float16* vp = vt + (size_t)bh * 64 * 1024;

  // Q A-fragments for d-chunks [0,32) and [32,64)
  Frag16 aq[2];
  {
    const _Float16* rp = qp + (size_t)(q0 + lh) * 768;
    aq[0].q[0] = *(const v4u*)(rp + kg);
    aq[0].q[1] = *(const v4u*)(rp + kg + 16);
    aq[1].q[0] = *(const v4u*)(rp + 32 + kg);
    aq[1].q[1] = *(const v4u*)(rp + 32 + kg + 16);
  }

  float mrow[8], lrow[8];
  v8f oacc[4];
#pragma unroll
  for (int r = 0; r < 8; ++r) { mrow[r] = -1e30f; lrow[r] = 0.0f; }
#pragma unroll
  for (int d = 0; d < 4; ++d) oacc[d] = (v8f){};

  for (int key0 = 0; key0 < 1024; key0 += 32) {
    // ---- S = (Q @ K^T) over 32 keys: two 16-key tiles ----
    v8f s0 = (v8f){}, s1 = (v8f){};
    {
      Frag16 bk;
      const _Float16* kr0 = kp + (size_t)(key0 + lh) * 768;
      bk.q[0] = *(const v4u*)(kr0 + kg);
      bk.q[1] = *(const v4u*)(kr0 + kg + 16);
      s0 = wmma_f16(aq[0].h, bk.h, s0);
      bk.q[0] = *(const v4u*)(kr0 + 32 + kg);
      bk.q[1] = *(const v4u*)(kr0 + 32 + kg + 16);
      s0 = wmma_f16(aq[1].h, bk.h, s0);
      const _Float16* kr1 = kp + (size_t)(key0 + 16 + lh) * 768;
      bk.q[0] = *(const v4u*)(kr1 + kg);
      bk.q[1] = *(const v4u*)(kr1 + kg + 16);
      s1 = wmma_f16(aq[0].h, bk.h, s1);
      bk.q[0] = *(const v4u*)(kr1 + 32 + kg);
      bk.q[1] = *(const v4u*)(kr1 + 32 + kg + 16);
      s1 = wmma_f16(aq[1].h, bk.h, s1);
    }

    // ---- online softmax over 32 keys (row = r + 8*(lane/16)) ----
#pragma unroll
    for (int r = 0; r < 8; ++r) {
      const float a0 = s0[r] * 0.125f;  // 64^-0.5
      const float a1 = s1[r] * 0.125f;
      float t = fmaxf(a0, a1);
      t = fmaxf(t, __shfl_xor(t, 1, 16));
      t = fmaxf(t, __shfl_xor(t, 2, 16));
      t = fmaxf(t, __shfl_xor(t, 4, 16));
      t = fmaxf(t, __shfl_xor(t, 8, 16));
      const float nm = fmaxf(mrow[r], t);
      const float corr = __expf(mrow[r] - nm);
      mrow[r] = nm;
      const float p0 = __expf(a0 - nm), p1 = __expf(a1 - nm);
      float rs = p0 + p1;
      rs += __shfl_xor(rs, 1, 16);
      rs += __shfl_xor(rs, 2, 16);
      rs += __shfl_xor(rs, 4, 16);
      rs += __shfl_xor(rs, 8, 16);
      lrow[r] = lrow[r] * corr + rs;
#pragma unroll
      for (int d = 0; d < 4; ++d) oacc[d][r] *= corr;
      const int prow = r + (lane >> 4) * 8;
      pbuf[wave][prow][lh]      = (_Float16)p0;
      pbuf[wave][prow][16 + lh] = (_Float16)p1;
    }

    // C-layout -> A-layout via per-wave LDS tile (same-wave DS ops are
    // in-order per ISA; no cross-wave sharing, so no barrier needed)
    Frag16 ap;
    ap.q[0] = *(const v4u*)(&pbuf[wave][lh][kg]);
    ap.q[1] = *(const v4u*)(&pbuf[wave][lh][kg + 16]);

    // ---- ctx += P @ V (V pre-transposed: keys contiguous) ----
#pragma unroll
    for (int d = 0; d < 4; ++d) {
      Frag16 bv;
      const _Float16* vr = vp + (size_t)(d * 16 + lh) * 1024 + key0 + kg;
      bv.q[0] = *(const v4u*)(vr);
      bv.q[1] = *(const v4u*)(vr + 16);
      oacc[d] = wmma_f16(ap.h, bv.h, oacc[d]);
    }
  }

  // ---- normalize + store ctx [token, 768] f16 ----
#pragma unroll
  for (int d = 0; d < 4; ++d)
#pragma unroll
    for (int r = 0; r < 8; ++r) {
      const int row = q0 + r + (lane >> 4) * 8;
      const float v = oacc[d][r] / lrow[r];
      ctx[((size_t)(b * 1024 + row)) * 768 + h * 64 + d * 16 + lh] =
          (_Float16)v;
    }
}

// ---------------------------------------------------------------------------
// Host launcher
// ---------------------------------------------------------------------------
extern "C" void kernel_launch(void* const* d_in, const int* in_sizes, int n_in,
                              void* d_out, int out_size, void* d_ws,
                              size_t ws_size, hipStream_t stream) {
  (void)in_sizes; (void)n_in; (void)out_size; (void)ws_size;
  const float* x     = (const float*)d_in[0];
  const float* ln1_g = (const float*)d_in[1];
  const float* ln1_b = (const float*)d_in[2];
  const float* Wq    = (const float*)d_in[3];
  const float* bq    = (const float*)d_in[4];
  const float* Wk    = (const float*)d_in[5];
  const float* bk    = (const float*)d_in[6];
  const float* Wv    = (const float*)d_in[7];
  const float* bv    = (const float*)d_in[8];
  const float* Wo    = (const float*)d_in[9];
  const float* bo    = (const float*)d_in[10];
  const float* ln2_g = (const float*)d_in[11];
  const float* ln2_b = (const float*)d_in[12];
  const float* W1    = (const float*)d_in[13];
  const float* b1    = (const float*)d_in[14];
  const float* W2    = (const float*)d_in[15];
  const float* b2    = (const float*)d_in[16];
  float* out = (float*)d_out;

  const int M = 16384;  // 16 * 1024 tokens

  char* w = (char*)d_ws;
  auto alloc = [&](size_t bytes) {
    char* p = w;
    w += (bytes + 255) & ~(size_t)255;
    return p;
  };
  _Float16* wq16 = (_Float16*)alloc((size_t)768 * 768 * 2);
  _Float16* wk16 = (_Float16*)alloc((size_t)768 * 768 * 2);
  _Float16* wv16 = (_Float16*)alloc((size_t)768 * 768 * 2);
  _Float16* wo16 = (_Float16*)alloc((size_t)768 * 768 * 2);
  _Float16* w116 = (_Float16*)alloc((size_t)768 * 3072 * 2);
  _Float16* w216 = (_Float16*)alloc((size_t)3072 * 768 * 2);
  _Float16* h16  = (_Float16*)alloc((size_t)M * 768 * 2);
  _Float16* q16  = (_Float16*)alloc((size_t)M * 768 * 2);
  _Float16* k16  = (_Float16*)alloc((size_t)M * 768 * 2);
  _Float16* vt16 = (_Float16*)alloc((size_t)M * 768 * 2);
  _Float16* ctx16 = (_Float16*)alloc((size_t)M * 768 * 2);
  float*    x1   = (float*)alloc((size_t)M * 768 * 4);
  _Float16* h216 = (_Float16*)alloc((size_t)M * 768 * 2);
  _Float16* g16  = (_Float16*)alloc((size_t)M * 3072 * 2);

  const int nw = 768 * 768, nf = 768 * 3072;
  cvt_kernel<<<(nw + 255) / 256, 256, 0, stream>>>(Wq, wq16, nw);
  cvt_kernel<<<(nw + 255) / 256, 256, 0, stream>>>(Wk, wk16, nw);
  cvt_kernel<<<(nw + 255) / 256, 256, 0, stream>>>(Wv, wv16, nw);
  cvt_kernel<<<(nw + 255) / 256, 256, 0, stream>>>(Wo, wo16, nw);
  cvt_kernel<<<(nf + 255) / 256, 256, 0, stream>>>(W1, w116, nf);
  cvt_kernel<<<(nf + 255) / 256, 256, 0, stream>>>(W2, w216, nf);

  ln_kernel<<<M, 256, 0, stream>>>(x, ln1_g, ln1_b, h16);

  dim3 g768(768 / 128, M / 128);
  dim3 g3072(3072 / 128, M / 128);
  gemm_kernel<0><<<g768, 256, 0, stream>>>(h16, wq16, bq, nullptr, q16,
                                           nullptr, M, 768, 768);
  gemm_kernel<0><<<g768, 256, 0, stream>>>(h16, wk16, bk, nullptr, k16,
                                           nullptr, M, 768, 768);
  gemm_kernel<1><<<g768, 256, 0, stream>>>(h16, wv16, bv, nullptr, vt16,
                                           nullptr, M, 768, 768);

  attn_kernel<<<16 * 12 * 8, 256, 0, stream>>>(q16, k16, vt16, ctx16);

  gemm_kernel<3><<<g768, 256, 0, stream>>>(ctx16, wo16, bo, x, nullptr, x1,
                                           M, 768, 768);

  ln_kernel<<<M, 256, 0, stream>>>(x1, ln2_g, ln2_b, h216);

  gemm_kernel<2><<<g3072, 256, 0, stream>>>(h216, w116, b1, nullptr, g16,
                                            nullptr, M, 3072, 768);
  gemm_kernel<3><<<g768, 256, 0, stream>>>(g16, w216, b2, x1, nullptr, out,
                                           M, 768, 3072);
}
